// MultiHead_Attention_Lattice_rel_save_gpumm_36893769073024
// MI455X (gfx1250) — matrix operand order repaired and checked by reading, not compiled, not run on hardware
//
#include <hip/hip_runtime.h>
#include <math.h>

// Problem constants (match reference)
#define B_   2
#define L_   256
#define H_   512
#define NH_  8
#define DH_  64
#define ML_  (B_*L_)     // 512 rows when (b,i) flattened

typedef unsigned int       u32;
typedef unsigned long long u64;
typedef __attribute__((ext_vector_type(16))) _Float16 v16h;
typedef __attribute__((ext_vector_type(8)))  float    v8f;
typedef __attribute__((ext_vector_type(4)))  u32      v4u;
typedef __attribute__((ext_vector_type(8)))  int      v8i;
typedef __attribute__((ext_vector_type(4)))  int      v4i;

#define WMMA_F16(acc, a, b) \
  (acc) = __builtin_amdgcn_wmma_f32_16x16x32_f16(false, (a), false, (b), (short)0, (acc), false, false)

// ---------------------------------------------------------------------------
// WMMA tile loaders (CDNA5 wave32 layouts, ISA 7.12.2)
// A 16x32 f16: lane = hi*16+m, m=lane&15; VGPR v, elem e -> K = (v>=4?16:0)+hi*8+2*(v&3)+e
// B 32x16 f16: lane = hi*16+n, n=lane&15; VGPR v, elem e -> K = (v>=4?16:0)+hi*8+2*(v&3)+e
// C  16x16 f32: lane n=lane&15; VGPR r -> M = r + (lane>>4)*8
// ---------------------------------------------------------------------------

__device__ __forceinline__ v16h wload_A(const float* base, int lda, int lane) {
  int m = lane & 15, hi = lane >> 4;
  const float* r = base + m * lda + hi * 8;
  v16h o;
#pragma unroll
  for (int v = 0; v < 8; ++v) {
    int kk = ((v & 4) ? 16 : 0) + 2 * (v & 3);
    float2 p = *(const float2*)(r + kk);
    o[2*v]   = (_Float16)p.x;
    o[2*v+1] = (_Float16)p.y;
  }
  return o;
}

// A tile with a per-K-column additive bias (for q+u / q+v)
__device__ __forceinline__ v16h wload_A_bias(const float* base, int lda,
                                             const float* bias, int lane) {
  int m = lane & 15, hi = lane >> 4;
  const float* r  = base + m * lda + hi * 8;
  const float* bb = bias + hi * 8;
  v16h o;
#pragma unroll
  for (int v = 0; v < 8; ++v) {
    int kk = ((v & 4) ? 16 : 0) + 2 * (v & 3);
    float2 p = *(const float2*)(r + kk);
    o[2*v]   = (_Float16)(p.x + bb[kk]);
    o[2*v+1] = (_Float16)(p.y + bb[kk+1]);
  }
  return o;
}

// B tile, K-dimension contiguous in memory (stride 1), column stride sn
__device__ __forceinline__ v16h wload_B_k1(const float* base, int sn, int lane) {
  int n = lane & 15, hi = lane >> 4;
  const float* p = base + hi * 8 + n * sn;
  v16h o;
#pragma unroll
  for (int v = 0; v < 8; ++v) {
    float2 q = *(const float2*)(p + ((v & 4) ? 16 : 0) + 2 * (v & 3));
    o[2*v]   = (_Float16)q.x;
    o[2*v+1] = (_Float16)q.y;
  }
  return o;
}

// B tile, generic strides: element (K,n) at base + K*sk + n*sn
__device__ __forceinline__ v16h wload_B_gen(const float* base, int sk, int sn, int lane) {
  int n = lane & 15, hi = lane >> 4;
  const float* p = base + hi * 8 * sk + n * sn;
  v16h o;
#pragma unroll
  for (int v = 0; v < 8; ++v) {
    o[2*v]   = (_Float16)p[(((v & 4) ? 16 : 0) + 2 * (v & 3))     * sk];
    o[2*v+1] = (_Float16)p[(((v & 4) ? 16 : 0) + 2 * (v & 3) + 1) * sk];
  }
  return o;
}

// ---------------------------------------------------------------------------
// TDM: issue one tensor_load_to_lds moving a 2-D tile of f32:
//   tile_dim0 = 32 (contiguous h), tile_dim1 = 256 (rows j), row stride 512
// into LDS at byte offset lds_off (contiguous [j][32] layout).
// Descriptor per CDNA5 ISA §8.3/8.4; 6-arg builtin form on this toolchain:
//   (uint32x4 g0, int32x8 g1, int32x4, int32x4, int32x8, i32 cpol)
// ---------------------------------------------------------------------------
__device__ __forceinline__ void tdm_load_rpe_slab(const float* gptr, u32 lds_off) {
  u64 ga = (u64)(uintptr_t)gptr;
  v4u g0;
  g0[0] = 1u;                                             // count=1, user mode
  g0[1] = lds_off;                                        // lds_addr (bytes)
  g0[2] = (u32)(ga & 0xFFFFFFFFu);                        // global_addr[31:0]
  g0[3] = (u32)((ga >> 32) & 0x01FFFFFFu) | 0x80000000u;  // addr[56:32] | type=2
  v8i g1;
  g1[0] = 0x00020000;                // workgroup_mask=0, data_size=2 (4B)
  g1[1] = (int)(512u << 16);         // tensor_dim0 = 512  (bits 79:48)
  g1[2] = (int)(256u << 16);         // tensor_dim1 = 256  (bits 111:80)
  g1[3] = (int)(32u  << 16);         // tile_dim0   = 32   (bits 127:112)
  g1[4] = 256;                       // tile_dim1 = 256, tile_dim2 = 0
  g1[5] = 512;                       // tensor_dim0_stride[31:0] = 512
  g1[6] = 0;                         // stride0[47:32]=0 | stride1[15:0]=0
  g1[7] = 2;                         // tensor_dim1_stride = 131072 (bits 47:16)
  v4i z4 = {0, 0, 0, 0};             // groups 2/3 unused (2-D tile)
  v8i z8 = {0, 0, 0, 0, 0, 0, 0, 0};
  __builtin_amdgcn_tensor_load_to_lds(g0, g1, z4, z4, z8, 0);
}

// ---------------------------------------------------------------------------
// Kernel 1/6: generic GEMM  C[M,N] = A[M,K] @ W[K,N] + bias[N]
// grid (N/64, M/16), block 128 (4 waves, each one 16x16 N-tile)
// ---------------------------------------------------------------------------
__global__ void gemm_bias_kernel(const float* __restrict__ A,
                                 const float* __restrict__ W,
                                 const float* __restrict__ bias,
                                 float* __restrict__ C, int K, int N) {
  int lane = threadIdx.x & 31, wave = threadIdx.x >> 5;
  int n0 = blockIdx.x * 64 + wave * 16;
  int m0 = blockIdx.y * 16;
  v8f acc = {};
  for (int kk = 0; kk < K; kk += 32) {
    v16h a = wload_A(A + m0 * K + kk, K, lane);
    v16h b = wload_B_gen(W + kk * N + n0, N, 1, lane);
    WMMA_F16(acc, a, b);
  }
  int n = lane & 15, hi = lane >> 4;
  float bn = bias[n0 + n];
#pragma unroll
  for (int r = 0; r < 8; ++r)
    C[(m0 + r + hi * 8) * N + n0 + n] = acc[r] + bn;
}

// ---------------------------------------------------------------------------
// Kernel 2: P[b,i,n,h] = sum_d (qp[b,i,nD+d] + v[n,d]) * Wr[h, nD+d]
// grid (H/64, ML/16, NH), block 128
// ---------------------------------------------------------------------------
__global__ void p_kernel(const float* __restrict__ qp,
                         const float* __restrict__ Wr,
                         const float* __restrict__ vvec,
                         float* __restrict__ P) {
  int lane = threadIdx.x & 31, wave = threadIdx.x >> 5;
  int h0 = blockIdx.x * 64 + wave * 16;
  int m0 = blockIdx.y * 16;
  int nh = blockIdx.z;
  v8f acc = {};
#pragma unroll
  for (int kk = 0; kk < DH_; kk += 32) {
    v16h a = wload_A_bias(qp + m0 * H_ + nh * DH_ + kk, H_, vvec + nh * DH_ + kk, lane);
    v16h b = wload_B_k1(Wr + h0 * H_ + nh * DH_ + kk, H_, lane);   // Wr[h, nD+k]
    WMMA_F16(acc, a, b);
  }
  int n = lane & 15, hi = lane >> 4;
#pragma unroll
  for (int r = 0; r < 8; ++r) {
    int m = m0 + r + hi * 8;                 // flattened (b,i)
    P[(m * NH_ + nh) * H_ + h0 + n] = acc[r];
  }
}

// Kernel 2b: cBD[b,n,i] = sum_d (qp + v) * br   (one thread per row)
__global__ void cbd_kernel(const float* __restrict__ qp,
                           const float* __restrict__ br,
                           const float* __restrict__ vvec,
                           float* __restrict__ cBD) {
  int tid = blockIdx.x * blockDim.x + threadIdx.x;   // ((b*NH+n)*L + i)
  if (tid >= B_ * NH_ * L_) return;
  int b = tid / (NH_ * L_);
  int rem = tid % (NH_ * L_);
  int n = rem / L_, i = rem % L_;
  const float* q = qp + (b * L_ + i) * H_ + n * DH_;
  const float* bb = br + n * DH_;
  const float* vv = vvec + n * DH_;
  float s = 0.f;
#pragma unroll 8
  for (int d = 0; d < DH_; ++d) s += (q[d] + vv[d]) * bb[d];
  cBD[tid] = s;
}

// ---------------------------------------------------------------------------
// Kernel 3a: AC scores[b,n,i,j] = (q+u) . k   grid (L/64, L/16, B*NH), block 128
// ---------------------------------------------------------------------------
__global__ void ac_kernel(const float* __restrict__ qp,
                          const float* __restrict__ kp,
                          const float* __restrict__ uvec,
                          float* __restrict__ sc) {
  int lane = threadIdx.x & 31, wave = threadIdx.x >> 5;
  int j0 = blockIdx.x * 64 + wave * 16;
  int i0 = blockIdx.y * 16;
  int bn = blockIdx.z;
  int b = bn / NH_, n = bn % NH_;
  const float* qa = qp + (b * L_ + i0) * H_ + n * DH_;
  const float* kb = kp + (b * L_ + j0) * H_ + n * DH_;
  v8f acc = {};
#pragma unroll
  for (int kk = 0; kk < DH_; kk += 32) {
    v16h a = wload_A_bias(qa + kk, H_, uvec + n * DH_ + kk, lane);
    v16h bm = wload_B_k1(kb + kk, H_, lane);   // B[d,j] = kp[j, nD+d]
    WMMA_F16(acc, a, bm);
  }
  int nn = lane & 15, hi = lane >> 4;
#pragma unroll
  for (int r = 0; r < 8; ++r)
    sc[(bn * L_ + i0 + r + hi * 8) * L_ + j0 + nn] = acc[r];
}

// ---------------------------------------------------------------------------
// Kernel 3b: BD accumulate — streams rel_pos_embedding exactly once via the
// Tensor Data Mover, double-buffered in LDS; P staged once in LDS as f16 in
// row-major [16][512] layout (rows 8..15 zero).
//   grid (ML) = one block per (b,i); block 512 = 16 waves, wave w -> j-tile w.
//   scores[b,n,i,j] += sum_h P[b,i,n,h] * rpe[b,i,j,h] + cBD[b,n,i]
// ---------------------------------------------------------------------------
#define RPE_SLAB_BYTES (32 * 256 * 4)   // 32 h x 256 j of f32 = 32 KB

__global__ void __launch_bounds__(512) bd_kernel(const float* __restrict__ P,
                                                 const float* __restrict__ rpe,
                                                 const float* __restrict__ cBD,
                                                 float* __restrict__ sc) {
  __shared__ __align__(16) unsigned char smem[2 * RPE_SLAB_BYTES + 16 * H_ * 2];
  float*    bufs = (float*)smem;                              // 2 x [256][32] f32
  _Float16* Ph   = (_Float16*)(smem + 2 * RPE_SLAB_BYTES);    // [16][512] f16

  int tid  = threadIdx.x;
  int lane = tid & 31, wave = tid >> 5;
  int bi = blockIdx.x;                 // b*L + i
  int b = bi / L_, i = bi % L_;

  // ---- stage P slice (8 x 512 f32 -> 16 x 512 f16, rows 8..15 zero) ----
  {
    const float* Pg = P + (size_t)bi * NH_ * H_;
    int n  = tid >> 5;                 // 0..15
    int h0 = (tid & 31) * 16;
#pragma unroll
    for (int e = 0; e < 16; ++e) {
      float x = (n < NH_) ? Pg[n * H_ + h0 + e] : 0.0f;
      Ph[n * H_ + h0 + e] = (_Float16)x;
    }
  }

  // ---- TDM prologue: fetch k-slab 0 into buffer 0 ----
  const float* rbase = rpe + (size_t)bi * L_ * H_;   // rpe[b,i,:,:]
  if (wave == 0) tdm_load_rpe_slab(rbase, 0);

  v8f acc = {};
  for (int t = 0; t < H_ / 32; ++t) {
    if (wave == 0) {
      if (t + 1 < H_ / 32) {
        tdm_load_rpe_slab(rbase + (t + 1) * 32, (u32)(((t + 1) & 1) * RPE_SLAB_BYTES));
        __builtin_amdgcn_s_wait_tensorcnt(1);   // slab t landed
      } else {
        __builtin_amdgcn_s_wait_tensorcnt(0);
      }
    }
    __syncthreads();                             // publish slab t

    // B tile from LDS: element (K = (v>=4?16:0)+hi*8+2*(v&3)+e, col j) at buf[(j0+j)*32 + K]
    int j0 = wave * 16;
    int nn = lane & 15, hi = lane >> 4;
    const float* pcol = bufs + (size_t)(t & 1) * (RPE_SLAB_BYTES / 4)
                        + (j0 + nn) * 32 + hi * 8;
    v16h bm;
#pragma unroll
    for (int v = 0; v < 8; ++v) {
      float2 q = *(const float2*)(pcol + ((v & 4) ? 16 : 0) + 2 * (v & 3));
      bm[2*v]   = (_Float16)q.x;
      bm[2*v+1] = (_Float16)q.y;
    }

    // A tile from pre-converted f16 P in LDS (no conversion in hot loop)
    union { v16h v; u32 w[8]; } a;
    {
      int m = lane & 15;
      const u32* pw = (const u32*)(Ph + m * H_ + t * 32 + hi * 8);
#pragma unroll
      for (int v = 0; v < 8; ++v)
        a.w[v] = pw[((v & 4) ? 8 : 0) + (v & 3)];
    }

    WMMA_F16(acc, a.v, bm);
    __syncthreads();                             // slab t consumed; safe to overwrite
  }

  int nn = lane & 15, hi = lane >> 4;
  if (hi == 0) {                                 // only M rows 0..7 = valid heads
    int j0 = wave * 16;
#pragma unroll
    for (int r = 0; r < 8; ++r) {
      int idx = ((b * NH_ + r) * L_ + i) * L_ + j0 + nn;
      sc[idx] += acc[r] + cBD[(b * NH_ + r) * L_ + i];
    }
  }
}

// ---------------------------------------------------------------------------
// Kernel 4: masked softmax over j. One wave per row; 8 waves/block.
// ---------------------------------------------------------------------------
__global__ void softmax_kernel(float* __restrict__ sc,
                               const int* __restrict__ seq_len,
                               const int* __restrict__ lex_num) {
  int lane = threadIdx.x & 31, wave = threadIdx.x >> 5;
  int row = blockIdx.x * 8 + wave;          // (b*NH+n)*L + i
  int b = row / (NH_ * L_);
  int vj = seq_len[b] + lex_num[0];
  float* rp = sc + (size_t)row * L_;
  const float scale = 0.125f;               // 1/sqrt(64)
  float vals[L_ / 32];
  float mx = -3.0e38f;
#pragma unroll
  for (int t = 0; t < L_ / 32; ++t) {
    int j = lane + 32 * t;
    float x = rp[j] * scale;
    if (j >= vj) x = -1.0e15f;
    vals[t] = x;
    mx = fmaxf(mx, x);
  }
#pragma unroll
  for (int off = 16; off >= 1; off >>= 1) mx = fmaxf(mx, __shfl_xor(mx, off, 32));
  float s = 0.f;
#pragma unroll
  for (int t = 0; t < L_ / 32; ++t) {
    float e = __expf(vals[t] - mx);
    vals[t] = e;
    s += e;
  }
#pragma unroll
  for (int off = 16; off >= 1; off >>= 1) s += __shfl_xor(s, off, 32);
  float inv = 1.0f / s;
#pragma unroll
  for (int t = 0; t < L_ / 32; ++t) rp[lane + 32 * t] = vals[t] * inv;
}

// ---------------------------------------------------------------------------
// Kernel 5: ctx[b,i,nD+d] = sum_j attn[b,n,i,j] * vp[b,j,nD+d]
// grid (L/16, B*NH), block 128 (4 waves cover the 4 d-tiles)
// ---------------------------------------------------------------------------
__global__ void attnv_kernel(const float* __restrict__ attn,
                             const float* __restrict__ vp,
                             float* __restrict__ ctx) {
  int lane = threadIdx.x & 31, wave = threadIdx.x >> 5;
  int d0 = wave * 16;
  int i0 = blockIdx.x * 16;
  int bn = blockIdx.y;
  int b = bn / NH_, n = bn % NH_;
  const float* Aa = attn + ((size_t)bn * L_ + i0) * L_;        // [16, 256]
  const float* Bb = vp + (size_t)b * L_ * H_ + n * DH_ + d0;   // B[j,d] stride H
  v8f acc = {};
  for (int kk = 0; kk < L_; kk += 32) {
    v16h a  = wload_A(Aa + kk, L_, lane);
    v16h bm = wload_B_gen(Bb + kk * H_, H_, 1, lane);
    WMMA_F16(acc, a, bm);
  }
  int nn = lane & 15, hi = lane >> 4;
#pragma unroll
  for (int r = 0; r < 8; ++r)
    ctx[(b * L_ + i0 + r + hi * 8) * H_ + n * DH_ + d0 + nn] = acc[r];
}

// ---------------------------------------------------------------------------
extern "C" void kernel_launch(void* const* d_in, const int* in_sizes, int n_in,
                              void* d_out, int out_size, void* d_ws, size_t ws_size,
                              hipStream_t stream) {
  (void)in_sizes; (void)n_in; (void)out_size; (void)ws_size;
  const float* key   = (const float*)d_in[0];
  const float* query = (const float*)d_in[1];
  const float* value = (const float*)d_in[2];
  const int*   seqln = (const int*)d_in[3];
  const int*   lexn  = (const int*)d_in[4];
  const float* rpe   = (const float*)d_in[7];
  const float* Wk = (const float*)d_in[8];  const float* bk = (const float*)d_in[9];
  const float* Wq = (const float*)d_in[10]; const float* bq = (const float*)d_in[11];
  const float* Wv = (const float*)d_in[12]; const float* bv = (const float*)d_in[13];
  const float* Wr = (const float*)d_in[14]; const float* br = (const float*)d_in[15];
  const float* uu = (const float*)d_in[16]; const float* vv = (const float*)d_in[17];
  const float* Wf = (const float*)d_in[18]; const float* bf = (const float*)d_in[19];
  float* out = (float*)d_out;

  float* ws  = (float*)d_ws;
  float* qp  = ws;                                   // [ML, H]
  float* kp  = qp + (size_t)ML_ * H_;                // [ML, H]
  float* vp  = kp + (size_t)ML_ * H_;                // [ML, H]
  float* P   = vp + (size_t)ML_ * H_;                // [ML, NH, H]
  float* cBD = P + (size_t)ML_ * NH_ * H_;           // [B*NH*L]
  float* sc  = cBD + (size_t)B_ * NH_ * L_;          // [B,NH,L,L]
  float* ctx = sc + (size_t)B_ * NH_ * L_ * L_;      // [ML, H]

  dim3 gProj(H_ / 64, ML_ / 16);
  gemm_bias_kernel<<<gProj, 128, 0, stream>>>(query, Wq, bq, qp, H_, H_);
  gemm_bias_kernel<<<gProj, 128, 0, stream>>>(key,   Wk, bk, kp, H_, H_);
  gemm_bias_kernel<<<gProj, 128, 0, stream>>>(value, Wv, bv, vp, H_, H_);

  p_kernel<<<dim3(H_ / 64, ML_ / 16, NH_), 128, 0, stream>>>(qp, Wr, vv, P);
  cbd_kernel<<<dim3((B_ * NH_ * L_) / 256), 256, 0, stream>>>(qp, br, vv, cBD);

  ac_kernel<<<dim3(L_ / 64, L_ / 16, B_ * NH_), 128, 0, stream>>>(qp, kp, uu, sc);
  bd_kernel<<<dim3(ML_), 512, 0, stream>>>(P, rpe, cBD, sc);

  softmax_kernel<<<dim3((B_ * NH_ * L_) / 8), 256, 0, stream>>>(sc, seqln, lexn);

  attnv_kernel<<<dim3(L_ / 16, B_ * NH_), 128, 0, stream>>>(sc, vp, ctx);

  gemm_bias_kernel<<<gProj, 128, 0, stream>>>(ctx, Wf, bf, out, H_, H_);
}
